// Block_86921548136965
// MI455X (gfx1250) — matrix operand order repaired and checked by reading, not compiled, not run on hardware
//
#include <hip/hip_runtime.h>

// ---- problem constants (from reference) ----
#define C_DIM 384
#define HEADS 6
#define HD    64
#define GPTS  512     // G
#define BATCH 64      // B
#define NL2   128     // N2
#define KNN   32      // K
#define HIDN  1536    // 4*DIM
#define MROWS (BATCH*GPTS)   // 32768

typedef __attribute__((ext_vector_type(16))) __bf16 v16bf;
typedef __attribute__((ext_vector_type(8)))  float  v8f;

union Frag {
  unsigned int u[8];
  v16bf v;
};

__device__ __forceinline__ unsigned short f2bf(float f) {
  unsigned int u = __float_as_uint(f);
  u += 0x7FFFu + ((u >> 16) & 1u);           // round-to-nearest-even
  return (unsigned short)(u >> 16);
}

__device__ __forceinline__ float bf2f(unsigned short h) {
  return __uint_as_float(((unsigned int)h) << 16);
}

__device__ __forceinline__ v8f wmma_bf16(const Frag& a, const Frag& b, v8f c) {
  // emits v_wmma_f32_16x16x32_bf16
  return __builtin_amdgcn_wmma_f32_16x16x32_bf16(
      /*neg_a=*/false, a.v, /*neg_b=*/false, b.v,
      /*c_mod=*/(short)0, c, /*reuse_a=*/false, /*reuse_b=*/false);
}

// A fragment: 16x32 bf16 tile (MxK) of row-major [*, ld] matrix at (m0, k0).
// Lane half selects K+16; VGPR j holds K pair {2j,2j+1} (contiguous, matches B).
__device__ __forceinline__ void load_a_frag(Frag& f, const unsigned short* base,
                                            int ld, int m0, int k0, int r, int half) {
  const unsigned short* p = base + (size_t)(m0 + r) * ld + k0 + (half << 4);
#pragma unroll
  for (int j = 0; j < 8; j++) {
    const int koff = 2 * j;
    f.u[j] = *(const unsigned int*)(p + koff);
  }
}

// B fragment: 32x16 bf16 tile (KxN) where B[kk][n] = W[n0+n][k0+kk], W row-major [N, ld].
// Lane r = column n; lane half selects K+16; VGPR j holds K pair {2j,2j+1}.
__device__ __forceinline__ void load_bt_frag(Frag& f, const unsigned short* base,
                                             int ld, int n0, int k0, int r, int half) {
  const unsigned short* p = base + (size_t)(n0 + r) * ld + k0 + (half << 4);
#pragma unroll
  for (int j = 0; j < 8; j++) f.u[j] = *(const unsigned int*)(p + 2 * j);
}

// ---------------- f32 -> bf16 conversion (weights) ----------------
__global__ __launch_bounds__(256) void cvt_bf16_kernel(const float* __restrict__ in,
                                                       unsigned short* __restrict__ out, int n) {
  int i = blockIdx.x * 256 + threadIdx.x;
  if (i < n) out[i] = f2bf(in[i]);
}

// ---------------- LayerNorm(384) -> bf16, one wave per row ----------------
__global__ __launch_bounds__(256) void ln_bf16_kernel(const float* __restrict__ x,
                                                      const float* __restrict__ gam,
                                                      const float* __restrict__ bet,
                                                      unsigned short* __restrict__ out) {
  const int wave = threadIdx.x >> 5;
  const int lane = threadIdx.x & 31;
  const int row  = blockIdx.x * 8 + wave;
  const float* xr = x + (size_t)row * C_DIM;
  float v[12];
  float s = 0.f;
#pragma unroll
  for (int i = 0; i < 12; i++) { v[i] = xr[lane + i * 32]; s += v[i]; }
#pragma unroll
  for (int off = 16; off > 0; off >>= 1) s += __shfl_xor(s, off, 32);
  const float mu = s * (1.0f / C_DIM);
  float vs = 0.f;
#pragma unroll
  for (int i = 0; i < 12; i++) { float d = v[i] - mu; vs += d * d; }
#pragma unroll
  for (int off = 16; off > 0; off >>= 1) vs += __shfl_xor(vs, off, 32);
  const float inv = rsqrtf(vs * (1.0f / C_DIM) + 1e-5f);
#pragma unroll
  for (int i = 0; i < 12; i++) {
    int c = lane + i * 32;
    out[(size_t)row * C_DIM + c] = f2bf((v[i] - mu) * inv * gam[c] + bet[c]);
  }
}

// ---------------- WMMA bf16 GEMM: C[M,N] = A[M,K] * W[N,K]^T ----------------
// WG = 256 thr = 8 waves; WG tile 64(M) x 128(N); each wave 32x32 (4 accumulators).
// EPI 0: scatter to Q/K/V [B,H,G,64] bf16
// EPI 1: out f32 = acc + bias[col] + resid[row,col]
// EPI 2: out bf16 = gelu(acc + bias[col])
template <int EPI>
__global__ __launch_bounds__(256) void gemm_bf16_kernel(
    const unsigned short* __restrict__ A, const unsigned short* __restrict__ W,
    const int M, const int N, const int Kd,
    const float* __restrict__ bias, const float* __restrict__ resid,
    float* __restrict__ outf, unsigned short* __restrict__ outb,
    unsigned short* __restrict__ qo, unsigned short* __restrict__ ko,
    unsigned short* __restrict__ vo) {
  const int wave = threadIdx.x >> 5;
  const int lane = threadIdx.x & 31;
  const int half = lane >> 4;
  const int r    = lane & 15;
  const int tilesN = N >> 7;
  const int m0 = ((blockIdx.x / tilesN) << 6) + ((wave >> 2) << 5);
  const int n0 = ((blockIdx.x % tilesN) << 7) + ((wave & 3) << 5);

  const v8f zero = {0.f, 0.f, 0.f, 0.f, 0.f, 0.f, 0.f, 0.f};
  v8f acc[2][2];
  acc[0][0] = zero; acc[0][1] = zero; acc[1][0] = zero; acc[1][1] = zero;

  for (int k0 = 0; k0 < Kd; k0 += 32) {
    Frag a0, a1, b0, b1;
    load_a_frag(a0, A, Kd, m0, k0, r, half);
    load_a_frag(a1, A, Kd, m0 + 16, k0, r, half);
    load_bt_frag(b0, W, Kd, n0, k0, r, half);
    load_bt_frag(b1, W, Kd, n0 + 16, k0, r, half);
    acc[0][0] = wmma_bf16(a0, b0, acc[0][0]);
    acc[0][1] = wmma_bf16(a0, b1, acc[0][1]);
    acc[1][0] = wmma_bf16(a1, b0, acc[1][0]);
    acc[1][1] = wmma_bf16(a1, b1, acc[1][1]);
  }

#pragma unroll
  for (int mi = 0; mi < 2; mi++)
#pragma unroll
    for (int ni = 0; ni < 2; ni++)
#pragma unroll
      for (int e = 0; e < 8; e++) {
        const int row = m0 + mi * 16 + e + (half << 3);
        const int col = n0 + ni * 16 + r;
        const float val = acc[mi][ni][e];
        if (EPI == 0) {
          const int which = col / C_DIM;
          const int rem = col - which * C_DIM;
          const int hh = rem >> 6;
          const int d  = rem & 63;
          const int bb = row >> 9;
          const int gg = row & 511;
          unsigned short* dst = (which == 0) ? qo : (which == 1 ? ko : vo);
          dst[(((size_t)(bb * HEADS + hh) << 9) + gg) * HD + d] = f2bf(val);
        } else if (EPI == 1) {
          const size_t o = (size_t)row * N + col;
          outf[o] = val + bias[col] + resid[o];
        } else {
          const float t = val + bias[col];
          const float g = 0.5f * t * (1.0f + erff(t * 0.70710678118f));
          outb[(size_t)row * N + col] = f2bf(g);
        }
      }
}

// ---------------- attention: per (b,h, 32-query tile) ----------------
__global__ __launch_bounds__(256) void attn_kernel(const unsigned short* __restrict__ qb,
                                                   const unsigned short* __restrict__ kb,
                                                   const unsigned short* __restrict__ vb,
                                                   unsigned short* __restrict__ ob) {
  __shared__ float S[32 * 512];           // 64 KB scores
  __shared__ unsigned short P[32 * 512];  // 32 KB probabilities (bf16)

  const int wave = threadIdx.x >> 5;
  const int lane = threadIdx.x & 31;
  const int half = lane >> 4;
  const int r    = lane & 15;
  const int bh = blockIdx.x >> 4;          // b*H + h
  const int qt = (blockIdx.x & 15) << 5;   // query tile base (32 rows)
  const int b = bh / HEADS;
  const int h = bh - b * HEADS;
  const unsigned short* Q  = qb + (size_t)bh * GPTS * HD;
  const unsigned short* Km = kb + (size_t)bh * GPTS * HD;
  const unsigned short* V  = vb + (size_t)bh * GPTS * HD;
  const v8f zero = {0.f, 0.f, 0.f, 0.f, 0.f, 0.f, 0.f, 0.f};

  // ---- phase 1: S = scale * Q K^T (32 x 512), wave owns 64 keys ----
  {
    const int n0 = wave << 6;
    v8f acc[2][4];
#pragma unroll
    for (int mi = 0; mi < 2; mi++)
#pragma unroll
      for (int ni = 0; ni < 4; ni++) acc[mi][ni] = zero;
    for (int k0 = 0; k0 < HD; k0 += 32) {
      Frag a0, a1, bf[4];
      load_a_frag(a0, Q, HD, qt, k0, r, half);
      load_a_frag(a1, Q, HD, qt + 16, k0, r, half);
#pragma unroll
      for (int ni = 0; ni < 4; ni++) load_bt_frag(bf[ni], Km, HD, n0 + ni * 16, k0, r, half);
#pragma unroll
      for (int ni = 0; ni < 4; ni++) {
        acc[0][ni] = wmma_bf16(a0, bf[ni], acc[0][ni]);
        acc[1][ni] = wmma_bf16(a1, bf[ni], acc[1][ni]);
      }
    }
    const float scale = 0.125f;  // 64^-0.5
#pragma unroll
    for (int mi = 0; mi < 2; mi++)
#pragma unroll
      for (int ni = 0; ni < 4; ni++)
#pragma unroll
        for (int e = 0; e < 8; e++)
          S[(mi * 16 + e + (half << 3)) * 512 + n0 + ni * 16 + r] = acc[mi][ni][e] * scale;
  }
  __syncthreads();

  // ---- phase 2: softmax, one wave per 4 rows ----
  for (int rr = 0; rr < 4; rr++) {
    const int row = wave * 4 + rr;
    const float* Sr = S + row * 512;
    float v[16];
    float mx = -3.0e38f;
#pragma unroll
    for (int i = 0; i < 16; i++) { v[i] = Sr[lane + i * 32]; mx = fmaxf(mx, v[i]); }
#pragma unroll
    for (int off = 16; off > 0; off >>= 1) mx = fmaxf(mx, __shfl_xor(mx, off, 32));
    float sum = 0.f;
#pragma unroll
    for (int i = 0; i < 16; i++) { v[i] = __expf(v[i] - mx); sum += v[i]; }
#pragma unroll
    for (int off = 16; off > 0; off >>= 1) sum += __shfl_xor(sum, off, 32);
    const float inv = 1.0f / sum;
#pragma unroll
    for (int i = 0; i < 16; i++) P[row * 512 + lane + i * 32] = f2bf(v[i] * inv);
  }
  __syncthreads();

  // ---- phase 3: O = P V, one 16x16 output tile per wave ----
  {
    const int mi = wave >> 2;           // query sub-tile
    const int d0 = (wave & 3) << 4;     // head-dim sub-tile
    v8f acc = zero;
    for (int k0 = 0; k0 < 512; k0 += 32) {
      Frag a, bf;
      const unsigned short* Pr = P + (mi * 16 + r) * 512 + k0 + (half << 4);
#pragma unroll
      for (int j = 0; j < 8; j++) {
        const int koff = 2 * j;
        a.u[j] = *(const unsigned int*)(Pr + koff);
      }
#pragma unroll
      for (int j = 0; j < 8; j++) {
        const int kk = k0 + (half << 4) + 2 * j;
        unsigned int lo = V[(size_t)kk * HD + d0 + r];
        unsigned int hi = V[(size_t)(kk + 1) * HD + d0 + r];
        bf.u[j] = lo | (hi << 16);
      }
      acc = wmma_bf16(a, bf, acc);
    }
#pragma unroll
    for (int e = 0; e < 8; e++) {
      const int g = qt + mi * 16 + e + (half << 3);
      const int c = h * HD + d0 + r;
      ob[((size_t)(b * GPTS + g)) * C_DIM + c] = f2bf(acc[e]);
    }
  }
}

// ---------------- neighbor max-pool + BN + GELU + 0.3*centers ----------------
__global__ __launch_bounds__(128) void pool_vis_kernel(
    const float* __restrict__ xfin, const int* __restrict__ idx1, const int* __restrict__ idx2,
    const float* __restrict__ bn_g, const float* __restrict__ bn_b,
    const float* __restrict__ bn_mean, const float* __restrict__ bn_var,
    float* __restrict__ vis) {
  const int bj = blockIdx.x;                 // b*N2 + j
  const int t = threadIdx.x;                 // 128 threads, 3 channels each
  const int* nb = idx1 + (size_t)bj * KNN;
  float mx0 = -3.0e38f, mx1 = -3.0e38f, mx2 = -3.0e38f;
  for (int kk = 0; kk < KNN; kk++) {
    const float* src = xfin + (size_t)nb[kk] * C_DIM;
    mx0 = fmaxf(mx0, 2.0f * src[t]);
    mx1 = fmaxf(mx1, 2.0f * src[t + 128]);
    mx2 = fmaxf(mx2, 2.0f * src[t + 256]);
  }
  const float* ctr = xfin + (size_t)idx2[bj] * C_DIM;
  float mx[3] = {mx0, mx1, mx2};
#pragma unroll
  for (int q = 0; q < 3; q++) {
    const int c = t + q * 128;
    float p = (mx[q] - bn_mean[c]) * rsqrtf(bn_var[c] + 1e-5f) * bn_g[c] + bn_b[c];
    float g = 0.5f * p * (1.0f + erff(p * 0.70710678118f));
    vis[(size_t)bj * C_DIM + c] = g + 0.3f * ctr[c];
  }
}

// ---------------- 3-NN inverse distance interpolation, add into out ----------------
__global__ __launch_bounds__(128) void interp_kernel(const float* __restrict__ l1c,
                                                     const float* __restrict__ l2c,
                                                     const float* __restrict__ vis,
                                                     float* __restrict__ out) {
  __shared__ float d2s[NL2];
  __shared__ int bidx[3];
  __shared__ float bw[3];
  const int bi = blockIdx.x;            // b*G + i
  const int b = bi >> 9;
  const int t = threadIdx.x;            // == N2 threads
  const float px = l1c[(size_t)bi * 3 + 0];
  const float py = l1c[(size_t)bi * 3 + 1];
  const float pz = l1c[(size_t)bi * 3 + 2];
  const float* c2 = l2c + (size_t)(b * NL2 + t) * 3;
  const float dx = px - c2[0], dy = py - c2[1], dz = pz - c2[2];
  d2s[t] = dx * dx + dy * dy + dz * dz;
  __syncthreads();
  if (t == 0) {
    float v0 = 3.0e38f, v1 = 3.0e38f, v2 = 3.0e38f;
    int i0 = 0, i1 = 0, i2 = 0;
    for (int j = 0; j < NL2; j++) {
      float v = d2s[j];
      if (v < v0)      { v2 = v1; i2 = i1; v1 = v0; i1 = i0; v0 = v; i0 = j; }
      else if (v < v1) { v2 = v1; i2 = i1; v1 = v; i1 = j; }
      else if (v < v2) { v2 = v; i2 = j; }
    }
    float w0 = 1.0f / (v0 + 1e-8f), w1 = 1.0f / (v1 + 1e-8f), w2 = 1.0f / (v2 + 1e-8f);
    float s = 1.0f / (w0 + w1 + w2);
    bidx[0] = i0; bidx[1] = i1; bidx[2] = i2;
    bw[0] = w0 * s; bw[1] = w1 * s; bw[2] = w2 * s;
  }
  __syncthreads();
  const float* v0p = vis + (size_t)(b * NL2 + bidx[0]) * C_DIM;
  const float* v1p = vis + (size_t)(b * NL2 + bidx[1]) * C_DIM;
  const float* v2p = vis + (size_t)(b * NL2 + bidx[2]) * C_DIM;
  const float w0 = bw[0], w1 = bw[1], w2 = bw[2];
  for (int c = t; c < C_DIM; c += 128)
    out[(size_t)bi * C_DIM + c] += w0 * v0p[c] + w1 * v1p[c] + w2 * v2p[c];
}

extern "C" void kernel_launch(void* const* d_in, const int* in_sizes, int n_in,
                              void* d_out, int out_size, void* d_ws, size_t ws_size,
                              hipStream_t stream) {
  const float* x      = (const float*)d_in[0];
  const float* l1c    = (const float*)d_in[1];
  const float* l2c    = (const float*)d_in[2];
  const float* ln1_g  = (const float*)d_in[3];
  const float* ln1_b  = (const float*)d_in[4];
  const float* qkv_w  = (const float*)d_in[5];
  const float* proj_w = (const float*)d_in[6];
  const float* proj_b = (const float*)d_in[7];
  const float* ln2_g  = (const float*)d_in[8];
  const float* ln2_b  = (const float*)d_in[9];
  const float* fc1_w  = (const float*)d_in[10];
  const float* fc1_b  = (const float*)d_in[11];
  const float* fc2_w  = (const float*)d_in[12];
  const float* fc2_b  = (const float*)d_in[13];
  const float* bn_g   = (const float*)d_in[14];
  const float* bn_b   = (const float*)d_in[15];
  const float* bn_mean= (const float*)d_in[16];
  const float* bn_var = (const float*)d_in[17];
  const int*  idx1    = (const int*)d_in[18];
  const int*  idx2    = (const int*)d_in[19];
  float* out = (float*)d_out;

  // workspace layout (256B aligned regions)
  char* w = (char*)d_ws;
  size_t off = 0;
  auto take = [&](size_t bytes) -> void* {
    void* p = w + off;
    off += (bytes + 255) & ~(size_t)255;
    return p;
  };
  unsigned short* wq  = (unsigned short*)take((size_t)3 * C_DIM * C_DIM * 2);   // 1152x384
  unsigned short* wp  = (unsigned short*)take((size_t)C_DIM * C_DIM * 2);       // 384x384
  unsigned short* w1  = (unsigned short*)take((size_t)HIDN * C_DIM * 2);        // 1536x384
  unsigned short* w2  = (unsigned short*)take((size_t)C_DIM * HIDN * 2);        // 384x1536
  unsigned short* hbf = (unsigned short*)take((size_t)MROWS * C_DIM * 2);       // LN out (reused LN2, vis)
  unsigned short* qb  = (unsigned short*)take((size_t)MROWS * C_DIM * 2);       // q bf16
  unsigned short* kb  = (unsigned short*)take((size_t)MROWS * C_DIM * 2);       // k bf16
  unsigned short* vb  = (unsigned short*)take((size_t)MROWS * C_DIM * 2);       // v bf16
  unsigned short* obf = (unsigned short*)take((size_t)MROWS * C_DIM * 2);       // attn out bf16
  float* x1f = (float*)take((size_t)MROWS * C_DIM * 4);                         // post-attn residual f32
  unsigned short* midbf = qb;        // alias: MLP mid [32768,1536] bf16 == qkv+o region
  float* visf = (float*)hbf;         // alias: vis [B*N2,384] f32 fits in hbf region

  // 1) weights -> bf16
  cvt_bf16_kernel<<<(3 * C_DIM * C_DIM + 255) / 256, 256, 0, stream>>>(qkv_w, wq, 3 * C_DIM * C_DIM);
  cvt_bf16_kernel<<<(C_DIM * C_DIM + 255) / 256, 256, 0, stream>>>(proj_w, wp, C_DIM * C_DIM);
  cvt_bf16_kernel<<<(HIDN * C_DIM + 255) / 256, 256, 0, stream>>>(fc1_w, w1, HIDN * C_DIM);
  cvt_bf16_kernel<<<(C_DIM * HIDN + 255) / 256, 256, 0, stream>>>(fc2_w, w2, C_DIM * HIDN);

  // 2) LN1 -> bf16
  ln_bf16_kernel<<<MROWS / 8, 256, 0, stream>>>(x, ln1_g, ln1_b, hbf);

  // 3) QKV GEMM (32768 x 1152 x 384), scatter to per-head layout
  gemm_bf16_kernel<0><<<(MROWS / 64) * (3 * C_DIM / 128), 256, 0, stream>>>(
      hbf, wq, MROWS, 3 * C_DIM, C_DIM, nullptr, nullptr, nullptr, nullptr, qb, kb, vb);

  // 4) attention: B*H*16 query tiles
  attn_kernel<<<BATCH * HEADS * (GPTS / 32), 256, 0, stream>>>(qb, kb, vb, obf);

  // 5) proj GEMM + bias + residual(x) -> x1 f32
  gemm_bf16_kernel<1><<<(MROWS / 64) * (C_DIM / 128), 256, 0, stream>>>(
      obf, wp, MROWS, C_DIM, C_DIM, proj_b, x, x1f, nullptr, nullptr, nullptr, nullptr);

  // 6) LN2 -> bf16 (reuses hbf)
  ln_bf16_kernel<<<MROWS / 8, 256, 0, stream>>>(x1f, ln2_g, ln2_b, hbf);

  // 7) FC1 GEMM + bias + GELU -> bf16 mid
  gemm_bf16_kernel<2><<<(MROWS / 64) * (HIDN / 128), 256, 0, stream>>>(
      hbf, w1, MROWS, HIDN, C_DIM, fc1_b, nullptr, nullptr, midbf, nullptr, nullptr, nullptr);

  // 8) FC2 GEMM + bias + residual(x1) -> out f32
  gemm_bf16_kernel<1><<<(MROWS / 64) * (C_DIM / 128), 256, 0, stream>>>(
      midbf, w2, MROWS, C_DIM, HIDN, fc2_b, x1f, out, nullptr, nullptr, nullptr, nullptr);

  // 9) neighbor max-pool + BN + GELU + 0.3*centers -> vis
  pool_vis_kernel<<<BATCH * NL2, 128, 0, stream>>>(out, idx1, idx2, bn_g, bn_b, bn_mean, bn_var, visf);

  // 10) 3-NN inverse-distance interpolation, add into out
  interp_kernel<<<MROWS, 128, 0, stream>>>(l1c, l2c, visf, out);
}
